// SideChainProteinFeatures_16947940950372
// MI455X (gfx1250) — matrix operand order repaired and checked by reading, not compiled, not run on hardware
//
#include <hip/hip_runtime.h>
#include <hip/hip_bf16.h>

#define B_ 2
#define L_ 1024
#define A_ 14
#define K_ 30
#define NUM_RBF_ 16
#define NPE_ 16
#define EF_ 128
#define EDGE_IN_ 3152          // NPE + NUM_RBF*A*A
#define KPAD_ 3168             // = 16 pos + 16 zero-gap + 3136 rbf = 99*32
#define NEDGE_ (B_*L_*K_)      // 61440
#define M_TILE_ 32             // edges per block (2 WMMA M-tiles)

typedef _Float16 v16h __attribute__((ext_vector_type(16)));
typedef float    v8f  __attribute__((ext_vector_type(8)));
typedef _Float16 h2   __attribute__((ext_vector_type(2)));

// ---------------------------------------------------------------------------
// Kernel 1: masked distances + top-K nearest neighbors (exact tie-break)
// ---------------------------------------------------------------------------
__global__ __launch_bounds__(256) void topk_kernel(const float* __restrict__ X,
                                                   const float* __restrict__ mask,
                                                   int* __restrict__ eidx_ws,
                                                   int* __restrict__ eidx_out) {
  const int row = blockIdx.x;            // b*L + i
  const int b = row / L_;
  const int t = threadIdx.x;
  __shared__ float shD[L_];
  __shared__ unsigned int shMaxBits;
  __shared__ unsigned long long shBest;

  const float xi0 = X[((size_t)row*A_ + 1)*3 + 0];
  const float xi1 = X[((size_t)row*A_ + 1)*3 + 1];
  const float xi2 = X[((size_t)row*A_ + 1)*3 + 2];
  const float mi  = mask[row];
  if (t == 0) shMaxBits = 0u;
  __syncthreads();

  for (int j = t; j < L_; j += 256) {
    const int rj = b*L_ + j;
    const float dx = xi0 - X[((size_t)rj*A_ + 1)*3 + 0];
    const float dy = xi1 - X[((size_t)rj*A_ + 1)*3 + 1];
    const float dz = xi2 - X[((size_t)rj*A_ + 1)*3 + 2];
    const float m2 = mi * mask[rj];
    const float D  = m2 * sqrtf(dx*dx + dy*dy + dz*dz + 1e-6f);
    shD[j] = D;
    atomicMax(&shMaxBits, __float_as_uint(D));   // D >= 0 -> bit order == numeric order
  }
  __syncthreads();
  const float Dmax = __uint_as_float(shMaxBits);
  for (int j = t; j < L_; j += 256) {
    const float m2 = mi * mask[b*L_ + j];
    shD[j] = shD[j] + 2.0f*(1.0f - m2)*Dmax;     // D_adjust
  }
  __syncthreads();

  for (int k = 0; k < K_; ++k) {
    if (t == 0) shBest = ~0ull;
    __syncthreads();
    unsigned long long best = ~0ull;
    for (int j = t; j < L_; j += 256) {
      const unsigned long long p =
          ((unsigned long long)__float_as_uint(shD[j]) << 32) | (unsigned int)j;
      best = (p < best) ? p : best;
    }
    atomicMin(&shBest, best);                    // smallest value, lowest index
    __syncthreads();
    const int jb = (int)(shBest & 0xffffffffull);
    if (t == 0) {
      eidx_ws[row*K_ + k]  = jb;
      eidx_out[row*K_ + k] = jb;
      shD[jb] = __builtin_inff();
    }
    __syncthreads();
  }
}

// ---------------------------------------------------------------------------
// Kernel 2: build X2 = [N, Ca, C, O, Cb(virtual), X[5:]]  (B,L,14,3)
// ---------------------------------------------------------------------------
__global__ __launch_bounds__(256) void x2_kernel(const float* __restrict__ X,
                                                 float* __restrict__ X2) {
  const int r = blockIdx.x*256 + threadIdx.x;    // b*L + i
  if (r >= B_*L_) return;
  const float* p = X  + (size_t)r*A_*3;
  float*       o = X2 + (size_t)r*A_*3;
  float Nv[3], Ca[3], Cv[3];
#pragma unroll
  for (int c = 0; c < 3; ++c) { Nv[c] = p[0*3+c]; Ca[c] = p[1*3+c]; Cv[c] = p[2*3+c]; }
  float bb[3], cc[3], aa[3];
#pragma unroll
  for (int c = 0; c < 3; ++c) { bb[c] = Ca[c] - Nv[c]; cc[c] = Cv[c] - Ca[c]; }
  aa[0] = bb[1]*cc[2] - bb[2]*cc[1];
  aa[1] = bb[2]*cc[0] - bb[0]*cc[2];
  aa[2] = bb[0]*cc[1] - bb[1]*cc[0];
#pragma unroll
  for (int c = 0; c < 3; ++c) {
    o[0*3+c] = Nv[c];
    o[1*3+c] = Ca[c];
    o[2*3+c] = Cv[c];
    o[3*3+c] = p[3*3+c];
    o[4*3+c] = -0.58273431f*aa[c] + 0.56802827f*bb[c] - 0.54067466f*cc[c] + Ca[c];
  }
  for (int a = 5; a < A_; ++a)
#pragma unroll
    for (int c = 0; c < 3; ++c) o[a*3+c] = p[a*3+c];
}

// ---------------------------------------------------------------------------
// Kernel 3: W_edge (128 x 3152, f32) -> Wt (128 x KPAD, f16, N-MAJOR)
//   Padded feature layout: [pos(16) | zeros(16) | rbf(3136)] so that chunks
//   1..98 of the K loop are pure RBF (branch-free staging).
// ---------------------------------------------------------------------------
__global__ __launch_bounds__(256) void wt_kernel(const float* __restrict__ W,
                                                 _Float16* __restrict__ Wt) {
  const int idx = blockIdx.x*256 + threadIdx.x;
  if (idx >= KPAD_*EF_) return;
  const int n = idx / KPAD_, f = idx % KPAD_;
  float v;
  if (f < NPE_)      v = W[(size_t)n*EDGE_IN_ + f];          // positional enc
  else if (f < 32)   v = 0.0f;                               // zero gap
  else               v = W[(size_t)n*EDGE_IN_ + (f - 16)];   // rbf block
  Wt[idx] = (_Float16)v;
}

// ---------------------------------------------------------------------------
// Kernel 4: fused feature build + WMMA GEMM + LayerNorm
//   block = 32 edges x 128 outputs, 8 waves; each wave: one 16-col N tile,
//   two M tiles -> two independent v_wmma per K-chunk (shared B fragment).
// ---------------------------------------------------------------------------
__global__ __launch_bounds__(256) void edge_gemm_kernel(
    const float* __restrict__ X2,
    const float* __restrict__ atom_mask,
    const int*   __restrict__ eidx,
    const _Float16* __restrict__ Wt,
    const float* __restrict__ gamma,
    const float* __restrict__ beta,
    float* __restrict__ Eout) {
  const int t    = threadIdx.x;
  const int wg   = blockIdx.x;
  const int lane = t & 31;
  const int wid  = t >> 5;                // wave id 0..7 -> N tile
  const int nloc = lane & 15;
  const int hi   = lane >> 4;

  __shared__ float shXi[M_TILE_][A_][3];
  __shared__ float shXj[M_TILE_][A_][3];
  __shared__ float shAmi[M_TILE_][A_];
  __shared__ float shAmj[M_TILE_][A_];
  __shared__ float shDpos[M_TILE_];
  __shared__ float shDab[M_TILE_*196];    // combo folded in; reused as C tile
  __shared__ _Float16 shA[M_TILE_*32];    // A tile: [edge][k]
  __shared__ _Float16 shB[128*48];        // B tile: [n][k], stride 48 halves
  __shared__ float shMean[M_TILE_];
  __shared__ float shRstd[M_TILE_];

  // ---- per-edge coordinates / masks (8 threads per edge) ----
  {
    const int m = t >> 3;                 // edge 0..31
    const int e = wg*M_TILE_ + m;
    const int b = e / (L_*K_);
    const int i = (e / K_) % L_;
    const int j = eidx[e];
    for (int a = t & 7; a < A_; a += 8) {
      const int ri = (b*L_ + i)*A_ + a;
      const int rj = (b*L_ + j)*A_ + a;
#pragma unroll
      for (int c = 0; c < 3; ++c) {
        shXi[m][a][c] = X2[(size_t)ri*3 + c];
        shXj[m][a][c] = X2[(size_t)rj*3 + c];
      }
      shAmi[m][a] = 1.0f - atom_mask[ri];
      shAmj[m][a] = 1.0f - atom_mask[rj];
    }
    if ((t & 7) == 0) shDpos[m] = (float)j - (float)i;
  }
  __syncthreads();

  // ---- 14x14 atom-pair distances; combo==0 -> sentinel so exp underflows
  //      to exactly +0.0 (matches exp(...) * 0 of the reference) ----
  for (int p = t; p < M_TILE_*196; p += 256) {
    const int m  = p / 196;
    const int q  = p % 196;
    const int a1 = q / A_;
    const int a2 = q % A_;
    const float dx = shXi[m][a1][0] - shXj[m][a2][0];
    const float dy = shXi[m][a1][1] - shXj[m][a2][1];
    const float dz = shXi[m][a1][2] - shXj[m][a2][2];
    const float d  = sqrtf(dx*dx + dy*dy + dz*dz + 1e-6f);
    const float combo = shAmi[m][a1] * shAmj[m][a2];  // in {0,1}
    shDab[p] = (combo > 0.5f) ? d : 1.0e7f;
  }
  __syncthreads();

  v8f acc0 = {}, acc1 = {};
  for (int c = 0; c < KPAD_/32; ++c) {
    const int f0 = c*32;
    // ---- stage A: 32 edges x 32 features ----
    if (c == 0) {
      // chunk 0: [pos(16) | zeros(16)]  (only chunk with divergent math)
#pragma unroll
      for (int r = 0; r < 4; ++r) {
        const int idx = t + r*256;        // 0..1023
        const int m  = idx >> 5;
        const int fl = idx & 31;
        float val = 0.0f;
        if (fl < NPE_) {
          const int p = fl & 7;
          const float freq = __expf(-(float)p * 1.1512925465f); // 10000^(-p/8)
          const float ang  = shDpos[m] * freq;
          val = (fl < 8) ? __cosf(ang) : __sinf(ang);
        }
        shA[idx] = (_Float16)val;
      }
    } else {
      // chunks 1..98: pure RBF, branch-free
#pragma unroll
      for (int r = 0; r < 4; ++r) {
        const int idx = t + r*256;        // 0..1023
        const int m  = idx >> 5;
        const int fl = idx & 31;
        const int pq = 2*(c-1) + (fl >> 4);   // a1*14 + a2
        const int rr = fl & 15;               // rbf index
        const float d = shDab[m*196 + pq];
        const float z = (d - (float)rr*(20.0f/15.0f)) * 0.8f;   // / sigma=1.25
        shA[idx] = (_Float16)__expf(-z*z);
      }
    }
    // ---- stage B: 128 n-rows x 32 k, n-major -> vectorized copy ----
    {
      const int n    = t >> 1;            // 0..127
      const int koff = (t & 1) * 16;      // 0 or 16
      const uint4* gp = (const uint4*)(Wt + (size_t)n*KPAD_ + f0 + koff);
      const uint4 w0 = gp[0];
      const uint4 w1 = gp[1];
      *(uint4*)(&shB[n*48 + koff + 0]) = w0;
      *(uint4*)(&shB[n*48 + koff + 8]) = w1;
    }
    __syncthreads();

    // ---- load fragments per CDNA5 VGPR layouts and accumulate ----
    v16h afrag0, afrag1, bfrag;
#pragma unroll
    for (int v = 0; v < 8; ++v) {
      // A (16-bit, 16x32): lanes0-15 K=0-7(v0-3)/16-23(v4-7); lanes16-31 +8
      const int ka = (v < 4 ? 2*v : 16 + 2*(v-4)) + (hi ? 8 : 0);
      const h2 pa0 = *(const h2*)(&shA[nloc*32 + ka]);        // edges 0..15
      const h2 pa1 = *(const h2*)(&shA[(16 + nloc)*32 + ka]); // edges 16..31
      afrag0[2*v] = pa0[0]; afrag0[2*v+1] = pa0[1];
      afrag1[2*v] = pa1[0]; afrag1[2*v+1] = pa1[1];
      // B (16-bit, 32x16): same K order as A (lanes0-15 K=0-7/16-23; lanes16-31 +8)
      const int kb = (v < 4 ? 2*v : 16 + 2*(v-4)) + (hi ? 8 : 0);
      const h2 pb = *(const h2*)(&shB[(wid*16 + nloc)*48 + kb]);
      bfrag[2*v] = pb[0]; bfrag[2*v+1] = pb[1];
    }
    acc0 = __builtin_amdgcn_wmma_f32_16x16x32_f16(
        false, afrag0, false, bfrag, (short)0, acc0, false, false);
    acc1 = __builtin_amdgcn_wmma_f32_16x16x32_f16(
        false, afrag1, false, bfrag, (short)0, acc1, false, false);
    __syncthreads();
  }

  // ---- write C tiles (reuse shDab region; Dab is dead now) ----
  float* shC = shDab;                     // 32*128 floats <= 32*196 floats
#pragma unroll
  for (int v = 0; v < 8; ++v) {
    const int m = v + (hi ? 8 : 0);       // C layout: VGPR v -> rows v / v+8
    shC[m*128 + wid*16 + nloc]        = acc0[v];
    shC[(16 + m)*128 + wid*16 + nloc] = acc1[v];
  }
  __syncthreads();

  // ---- LayerNorm over the 128 outputs of each edge ----
  if (t < M_TILE_) {
    float s = 0.0f, s2 = 0.0f;
    for (int n = 0; n < 128; ++n) { const float x = shC[t*128 + n]; s += x; s2 += x*x; }
    const float mean = s * (1.0f/128.0f);
    const float var  = s2 * (1.0f/128.0f) - mean*mean;
    shMean[t] = mean;
    shRstd[t] = rsqrtf(var + 1e-5f);
  }
  __syncthreads();
  {
    const int m  = t >> 3;                // edge 0..31 (8 threads per edge)
    const int nb = t & 7;                 // 16 outputs per thread
    const int e  = wg*M_TILE_ + m;
    const float mean = shMean[m], rstd = shRstd[m];
#pragma unroll
    for (int q = 0; q < 16; ++q) {
      const int n = nb*16 + q;
      Eout[(size_t)e*EF_ + n] = (shC[m*128 + n] - mean)*rstd*gamma[n] + beta[n];
    }
  }
}

// ---------------------------------------------------------------------------
extern "C" void kernel_launch(void* const* d_in, const int* in_sizes, int n_in,
                              void* d_out, int out_size, void* d_ws, size_t ws_size,
                              hipStream_t stream) {
  const float* X         = (const float*)d_in[0];
  const float* mask      = (const float*)d_in[1];
  // d_in[2] residue_idx, d_in[3] chain_labels: unused by the reference
  const float* atom_mask = (const float*)d_in[4];
  const float* W_edge    = (const float*)d_in[5];
  const float* ln_gamma  = (const float*)d_in[6];
  const float* ln_beta   = (const float*)d_in[7];

  float* Eout = (float*)d_out;
  int*   Eidx_out = (int*)d_out + (size_t)NEDGE_*EF_;  // second tuple output

  char* ws = (char*)d_ws;
  int*      eidx_ws = (int*)ws;                                  // 240 KB
  float*    X2      = (float*)(ws + 256*1024);                   // 336 KB
  _Float16* Wt      = (_Float16*)(ws + 256*1024 + 512*1024);     // 792 KB

  topk_kernel<<<B_*L_, 256, 0, stream>>>(X, mask, eidx_ws, Eidx_out);
  x2_kernel<<<(B_*L_ + 255)/256, 256, 0, stream>>>(X, X2);
  wt_kernel<<<(KPAD_*EF_ + 255)/256, 256, 0, stream>>>(W_edge, Wt);
  edge_gemm_kernel<<<NEDGE_/M_TILE_, 256, 0, stream>>>(X2, atom_mask, eidx_ws, Wt,
                                                       ln_gamma, ln_beta, Eout);
}
